// ScaledDotProductAttention_17171279250118
// MI455X (gfx1250) — matrix operand order repaired and checked by reading, not compiled, not run on hardware
//
#include <hip/hip_runtime.h>

// ---------------------------------------------------------------------------
// relu15-attention for MI455X (gfx1250, wave32, WMMA)
//   attn = clamp((Q/T) K^T, 0, 15)   [B,H,S,S]  fp32 (output #2)
//   out  = attn V                    [B,H,S,D]  fp32 (output #1)
// B=8 H=16 S=1024 D=128.
// HBM floor ~872MB @ 23.3TB/s => ~37us. f16 WMMA (16x16x32, f32 accum) keeps
// compute under that. K/V tiles are staged once per workgroup into LDS
// (double-buffered) so L2 traffic stays ~8x below the naive per-wave reads.
// attn/out are streamed with non-temporal stores to keep K/V L2-resident.
// ---------------------------------------------------------------------------

typedef _Float16 v16h __attribute__((ext_vector_type(16)));
typedef _Float16 v8h  __attribute__((ext_vector_type(8)));
typedef float    v8f  __attribute__((ext_vector_type(8)));
typedef float    v4f  __attribute__((ext_vector_type(4)));

#define CAT16(lo, hi) __builtin_shufflevector(lo, hi, 0,1,2,3,4,5,6,7,8,9,10,11,12,13,14,15)

#define NB  8
#define NH  16
#define NS  1024
#define ND  128
#define NQTOT ((size_t)NB * NH * NS * ND)           // 16,777,216 elements
#define INV_T 0.08838834764831845f                  // 1/sqrt(128)

#define KSTRIDE 136   // halves per K-stage row (16B aligned, conflict-friendly)
#define VSTRIDE 40    // halves per V-stage / scratch row

#define WMMA_F16(a, b, c) \
    __builtin_amdgcn_wmma_f32_16x16x32_f16(false, (a), false, (b), (short)0, (c), false, false)

// --- prep: V[bh][t][d] fp32 -> Vt[bh][d][t] f16, 64x64 LDS tile transpose --
__global__ __launch_bounds__(256) void transpose_v_kernel(const float* __restrict__ v,
                                                          _Float16* __restrict__ vt) {
    __shared__ float tile[64][65];
    const int bh = blockIdx.z;
    const int t0 = blockIdx.y * 64;
    const int d0 = blockIdx.x * 64;
    const float* vb = v + ((size_t)bh * NS + t0) * ND + d0;
#pragma unroll
    for (int it = 0; it < 16; ++it) {
        int i = threadIdx.x + it * 256;
        int tr = i >> 6, td = i & 63;
        tile[tr][td] = vb[tr * ND + td];
    }
    __syncthreads();
    _Float16* ob = vt + ((size_t)bh * ND + d0) * NS + t0;
#pragma unroll
    for (int it = 0; it < 16; ++it) {
        int i = threadIdx.x + it * 256;
        int od = i >> 6, ot = i & 63;
        ob[(size_t)od * NS + ot] = (_Float16)tile[ot][od];   // 128B-contig rows
    }
}

// --- fused attention: 8 waves/block, 16 Q rows per wave, shared K/V tiles ---
__global__ __launch_bounds__(256) void attn_kernel(const float* __restrict__ q,
                                                   const float* __restrict__ k,
                                                   const _Float16* __restrict__ vt,
                                                   float* __restrict__ out,
                                                   float* __restrict__ attn) {
    // double-buffered block-shared tiles + per-wave D->A relayout scratch
    __shared__ _Float16 kst[2][32 * KSTRIDE];    // K tile: 32 t-rows x 128 d (f16)
    __shared__ _Float16 vst[2][128 * VSTRIDE];   // Vt tile: 128 d-rows x 32 t (f16)
    __shared__ _Float16 sw_all[8 * 16 * VSTRIDE];

    const int tid  = threadIdx.x;
    const int lane = tid & 31;
    const int wave = tid >> 5;
    const int g    = lane >> 4;          // lane half (ISA K/M-group select)
    const int ln   = lane & 15;
    const int bh   = blockIdx.x >> 3;
    const int row0 = ((blockIdx.x & 7) << 7) + wave * 16;
    _Float16* sw = sw_all + wave * (16 * VSTRIDE);

    const float*    kg = k  + (size_t)bh * NS * ND;
    const _Float16* vg = vt + (size_t)bh * ND * NS;

    // staging assignment (block-wide): K 32x128 fp32 -> f16, V 128x32 f16 copy
    const int krow = tid >> 3;           // 0..31
    const int kcol = (tid & 7) * 16;     // 0..112 (16 halves per thread)
    const int vrow = tid >> 1;           // 0..127
    const int voff = (tid & 1) * 16;     // 0 or 16

    // ---- Q A-fragments: fp32 global -> f16 registers, pre-scaled by 1/T ----
    // 16-bit A layout: lane=M, VGPR0-3 hold K in [8g,8g+8), VGPR4-7 in [16+8g,..)
    const float* qb = q + ((size_t)bh * NS + row0 + ln) * ND;
    v16h qf[4];
#pragma unroll
    for (int c = 0; c < 4; ++c) {
        v4f a0 = *(const v4f*)(qb + c * 32 + 8 * g);
        v4f a1 = *(const v4f*)(qb + c * 32 + 8 * g + 4);
        v4f b0 = *(const v4f*)(qb + c * 32 + 16 + 8 * g);
        v4f b1 = *(const v4f*)(qb + c * 32 + 16 + 8 * g + 4);
        v16h f;
#pragma unroll
        for (int j = 0; j < 4; ++j) {
            f[j]      = (_Float16)(a0[j] * INV_T);
            f[j + 4]  = (_Float16)(a1[j] * INV_T);
            f[j + 8]  = (_Float16)(b0[j] * INV_T);
            f[j + 12] = (_Float16)(b1[j] * INV_T);
        }
        qf[c] = f;
    }

    v8f acc[8];
#pragma unroll
    for (int i = 0; i < 8; ++i) acc[i] = (v8f){0.f, 0.f, 0.f, 0.f, 0.f, 0.f, 0.f, 0.f};

    float* attn_b = attn + (size_t)bh * NS * NS + (size_t)row0 * NS;

    // ---- prologue: stage tile 0 into buffer 0 ----
    {
        const float* ks = kg + (size_t)krow * ND + kcol;
        v4f k0 = *(const v4f*)(ks), k1 = *(const v4f*)(ks + 4);
        v4f k2 = *(const v4f*)(ks + 8), k3 = *(const v4f*)(ks + 12);
        v8h h0, h1;
#pragma unroll
        for (int j = 0; j < 4; ++j) {
            h0[j] = (_Float16)k0[j]; h0[j + 4] = (_Float16)k1[j];
            h1[j] = (_Float16)k2[j]; h1[j + 4] = (_Float16)k3[j];
        }
        *(v8h*)(&kst[0][krow * KSTRIDE + kcol])     = h0;
        *(v8h*)(&kst[0][krow * KSTRIDE + kcol + 8]) = h1;
        const _Float16* vs = vg + (size_t)vrow * NS + voff;
        *(v8h*)(&vst[0][vrow * VSTRIDE + voff])     = *(const v8h*)(vs);
        *(v8h*)(&vst[0][vrow * VSTRIDE + voff + 8]) = *(const v8h*)(vs + 8);
    }
    __syncthreads();

    for (int ct = 0; ct < 32; ++ct) {
        const int buf = ct & 1;
        const int t0  = ct * 32;
        const bool has_next = (ct < 31);

        // issue next tile's global loads early (hide latency under compute)
        v4f nk0 = {}, nk1 = {}, nk2 = {}, nk3 = {};
        v8h nv0 = {}, nv1 = {};
        if (has_next) {
            const float* ks = kg + (size_t)(t0 + 32 + krow) * ND + kcol;
            nk0 = *(const v4f*)(ks);     nk1 = *(const v4f*)(ks + 4);
            nk2 = *(const v4f*)(ks + 8); nk3 = *(const v4f*)(ks + 12);
            const _Float16* vs = vg + (size_t)vrow * NS + t0 + 32 + voff;
            nv0 = *(const v8h*)(vs);     nv1 = *(const v8h*)(vs + 8);
        }

        // ---- GEMM1: S = Q_tile x K^T (two 16-wide N tiles) from LDS ----
        // 16-bit B layout: lane=N, K in [8g,8g+8) then [16+8g,16+8g+8)
        const _Float16* k0l = &kst[buf][ln * KSTRIDE + 8 * g];
        const _Float16* k1l = k0l + 16 * KSTRIDE;
        v8f s0 = (v8f){0.f, 0.f, 0.f, 0.f, 0.f, 0.f, 0.f, 0.f};
        v8f s1 = (v8f){0.f, 0.f, 0.f, 0.f, 0.f, 0.f, 0.f, 0.f};
#pragma unroll
        for (int c = 0; c < 4; ++c) {
            v8h l0 = *(const v8h*)(k0l + c * 32);
            v8h h0 = *(const v8h*)(k0l + c * 32 + 16);
            s0 = WMMA_F16(qf[c], CAT16(l0, h0), s0);
            v8h l1 = *(const v8h*)(k1l + c * 32);
            v8h h1 = *(const v8h*)(k1l + c * 32 + 16);
            s1 = WMMA_F16(qf[c], CAT16(l1, h1), s1);
        }

        // ---- relu15, stream fp32 attn (non-temporal), stash f16 in LDS ----
#pragma unroll
        for (int r = 0; r < 8; ++r) {
            float a0 = fminf(fmaxf(s0[r], 0.f), 15.f);
            float a1 = fminf(fmaxf(s1[r], 0.f), 15.f);
            int m = r + 8 * g;                       // C/D layout: lane=N, M=r+8g
            __builtin_nontemporal_store(a0, &attn_b[(size_t)m * NS + t0 + ln]);
            __builtin_nontemporal_store(a1, &attn_b[(size_t)m * NS + t0 + 16 + ln]);
            sw[m * VSTRIDE + ln]      = (_Float16)a0;
            sw[m * VSTRIDE + 16 + ln] = (_Float16)a1;
        }
        asm volatile("s_wait_dscnt 0x0" ::: "memory");   // wave-synchronous LDS
        v8h al = *(const v8h*)(sw + ln * VSTRIDE + 8 * g);
        v8h ah = *(const v8h*)(sw + ln * VSTRIDE + 16 + 8 * g);
        v16h af = CAT16(al, ah);

        // ---- GEMM2: acc[nt] += attn_tile x V_chunk from LDS Vt tile ----
        const _Float16* vl = &vst[buf][ln * VSTRIDE + 8 * g];
#pragma unroll
        for (int nt = 0; nt < 8; ++nt) {
            const _Float16* p = vl + nt * 16 * VSTRIDE;
            v8h bl  = *(const v8h*)(p);
            v8h bhh = *(const v8h*)(p + 16);
            acc[nt] = WMMA_F16(af, CAT16(bl, bhh), acc[nt]);
        }

        // ---- commit next tile into the other buffer ----
        if (has_next) {
            v8h h0, h1;
#pragma unroll
            for (int j = 0; j < 4; ++j) {
                h0[j] = (_Float16)nk0[j]; h0[j + 4] = (_Float16)nk1[j];
                h1[j] = (_Float16)nk2[j]; h1[j + 4] = (_Float16)nk3[j];
            }
            *(v8h*)(&kst[buf ^ 1][krow * KSTRIDE + kcol])     = h0;
            *(v8h*)(&kst[buf ^ 1][krow * KSTRIDE + kcol + 8]) = h1;
            *(v8h*)(&vst[buf ^ 1][vrow * VSTRIDE + voff])     = nv0;
            *(v8h*)(&vst[buf ^ 1][vrow * VSTRIDE + voff + 8]) = nv1;
        }
        __syncthreads();
    }

    // ---- write out tile (non-temporal stream) ----
    float* ob = out + ((size_t)bh * NS + row0) * ND;
#pragma unroll
    for (int nt = 0; nt < 8; ++nt) {
#pragma unroll
        for (int r = 0; r < 8; ++r) {
            int m = r + 8 * g;
            __builtin_nontemporal_store(acc[nt][r], &ob[(size_t)m * ND + nt * 16 + ln]);
        }
    }
}

extern "C" void kernel_launch(void* const* d_in, const int* in_sizes, int n_in,
                              void* d_out, int out_size, void* d_ws, size_t ws_size,
                              hipStream_t stream) {
    const float* q = (const float*)d_in[0];
    const float* k = (const float*)d_in[1];
    const float* v = (const float*)d_in[2];
    float* out  = (float*)d_out;                  // [B,H,S,D]
    float* attn = out + NQTOT;                    // [B,H,S,S]

    _Float16* vt = (_Float16*)d_ws;               // [B*H][D][S] f16 (~33.5MB)

    transpose_v_kernel<<<dim3(ND / 64, NS / 64, NB * NH), 256, 0, stream>>>(v, vt);
    attn_kernel<<<NB * NH * (NS / 128), 256, 0, stream>>>(q, k, vt, out, attn);
}
